// LinearAttention_36051955482999
// MI455X (gfx1250) — compile-verified, matrix operand order checked by reading, not run on hardware
//
#include <hip/hip_runtime.h>
#include <hip/hip_bf16.h>

typedef __attribute__((ext_vector_type(16))) __bf16 v16bf;
typedef __attribute__((ext_vector_type(8)))  float  v8f;

union FragBF {
    unsigned int u[8];
    v16bf v;
};
union U4H8 {
    uint4 u4;
    unsigned short h[8];
};

// native bf16 converts -> v_cvt_pk_bf16_f32 class hardware ops
__device__ inline unsigned short f32_to_bf16(float f) {
    union { __bf16 b; unsigned short u; } c;
    c.b = (__bf16)f;
    return c.u;
}
__device__ inline float bf16_to_f32(unsigned short h) {
    union { __bf16 b; unsigned short u; } c;
    c.u = h;
    return (float)c.b;
}
__device__ inline v8f v8f_zero() {
    v8f z;
#pragma unroll
    for (int i = 0; i < 8; ++i) z[i] = 0.0f;
    return z;
}

// ---- CDNA5 async copy: global -> LDS, 16 bytes per lane, tracked by ASYNCcnt ----
__device__ inline void async_b128(void* lds_ptr, const void* gptr) {
    unsigned int lds = (unsigned int)(unsigned long long)lds_ptr;  // addr[31:0] = LDS offset
    unsigned long long ga = (unsigned long long)gptr;
    asm volatile("global_load_async_to_lds_b128 %0, %1, off"
                 :: "v"(lds), "v"(ga) : "memory");
}
template <int N>
__device__ inline void wait_async() {
    asm volatile("s_wait_asynccnt %0" :: "i"(N) : "memory");
}

// Fragment load from row-major LDS tile [rows][LDK] (halfs), K contiguous.
// frag[lane][v] = T[row][kmap(v, lane>>4)], two consecutive K per dword:
//   v<4 : k = 2v + 8*half ; v>=4 : k = 16 + 2(v-4) + 8*half
template <int LDK>
__device__ inline void load_frag_rows(const unsigned short* t, int row, int kbase, FragBF& f) {
    const int half = ((threadIdx.x & 31) >> 4);
    const unsigned int* p = (const unsigned int*)(t + row * LDK + kbase);
    const int o = half * 4;
    f.u[0] = p[o + 0]; f.u[1] = p[o + 1]; f.u[2] = p[o + 2]; f.u[3] = p[o + 3];
    f.u[4] = p[o + 8]; f.u[5] = p[o + 9]; f.u[6] = p[o + 10]; f.u[7] = p[o + 11];
}

__device__ inline void store_out(float* p, float v) { *p = v; }
__device__ inline void store_out(unsigned short* p, float v) { *p = f32_to_bf16(v); }

// ============================================================
// f32 -> bf16 bulk conversion (bandwidth bound, one pass)
// ============================================================
__global__ __launch_bounds__(256) void cvt_bf16_kernel(const float* __restrict__ in,
                                                       unsigned short* __restrict__ out,
                                                       size_t n4) {
    size_t i = ((size_t)blockIdx.x * 256 + threadIdx.x);
    if (i < n4) {
        float4 x = *(const float4*)(in + i * 4);
        ushort4 o;
        o.x = f32_to_bf16(x.x); o.y = f32_to_bf16(x.y);
        o.z = f32_to_bf16(x.z); o.w = f32_to_bf16(x.w);
        *(ushort4*)(out + i * 4) = o;
    }
}

// ============================================================
// GEMM: Y[M][Nn] = act(X[M][K] @ W[Nn][K]^T + bias), X/W bf16.
// 128x128 tile/WG, 8 waves (wave: 32x64 = 2x4 WMMA tiles), BK=64,
// double-buffered async global->LDS staging.
// ============================================================
template <int ACT, typename TOUT>
__global__ __launch_bounds__(256) void proj_kernel(const unsigned short* __restrict__ X,
                                                   const unsigned short* __restrict__ W,
                                                   const float* __restrict__ bias,
                                                   TOUT* __restrict__ Y,
                                                   int M, int Nn, int K) {
    __shared__ __align__(16) unsigned short At[2][128 * 64];  // 16KB each
    __shared__ __align__(16) unsigned short Bt[2][128 * 64];

    const int t = threadIdx.x;
    const int wave = t >> 5;
    const int lane = t & 31;
    const int bm = blockIdx.x * 128;
    const int bn = blockIdx.y * 128;
    const int wm = (wave >> 1) * 32;  // 0,32,64,96
    const int wn = (wave & 1) * 64;   // 0,64

    v8f acc[2][4];
#pragma unroll
    for (int im = 0; im < 2; ++im)
#pragma unroll
        for (int in = 0; in < 4; ++in) acc[im][in] = v8f_zero();

    const int nk = K >> 6;  // K/64 tiles

    auto issue = [&](int ki, int buf) {
#pragma unroll
        for (int i = 0; i < 4; ++i) {
            int flat = t + 256 * i;        // 1024 chunks of 8 halfs
            int row = flat >> 3, q = flat & 7;
            async_b128(&At[buf][row * 64 + q * 8],
                       X + (size_t)(bm + row) * K + ki * 64 + q * 8);
            async_b128(&Bt[buf][row * 64 + q * 8],
                       W + (size_t)(bn + row) * K + ki * 64 + q * 8);
        }
    };

    issue(0, 0);
    for (int ki = 0; ki < nk; ++ki) {
        const int buf = ki & 1;
        if (ki + 1 < nk) {
            issue(ki + 1, buf ^ 1);
            wait_async<8>();   // the 8 just-issued may stay outstanding
        } else {
            wait_async<0>();
        }
        __syncthreads();

#pragma unroll
        for (int kk = 0; kk < 64; kk += 32) {
            FragBF a[2], b[4];
#pragma unroll
            for (int im = 0; im < 2; ++im)
                load_frag_rows<64>(At[buf], wm + im * 16 + (lane & 15), kk, a[im]);
#pragma unroll
            for (int in = 0; in < 4; ++in)
                load_frag_rows<64>(Bt[buf], wn + in * 16 + (lane & 15), kk, b[in]);
#pragma unroll
            for (int im = 0; im < 2; ++im)
#pragma unroll
                for (int in = 0; in < 4; ++in)
                    acc[im][in] = __builtin_amdgcn_wmma_f32_16x16x32_bf16(
                        false, a[im].v, false, b[in].v, (short)0, acc[im][in], false, false);
        }
        __syncthreads();
    }

    const int cn = lane & 15;
    const int ch = lane >> 4;
#pragma unroll
    for (int im = 0; im < 2; ++im)
#pragma unroll
        for (int in = 0; in < 4; ++in) {
            int gn = bn + wn + in * 16 + cn;
            float bv = bias[gn];
#pragma unroll
            for (int r = 0; r < 8; ++r) {
                int gm = bm + wm + im * 16 + r + 8 * ch;
                float y = acc[im][in][r] + bv;
                if (ACT == 1) y = (y > 0.0f) ? (y + 1.0f) : __expf(y);
                store_out(Y + (size_t)gm * Nn + gn, y);
            }
        }
}

// ============================================================
// kv[b,h][m][d] = sum_s v[s,m]*k[s,d]  and  ksum[d] = sum_s k[s,d]
// one WG per (b,h); tiles staged TRANSPOSED into LDS so all WMMA
// fragment loads are contiguous dword pairs. ksum is computed on the
// matrix pipe: a constant all-ones A fragment against the K fragments
// makes accumulator row M=0 the running k_sum (bf16 output).
// ============================================================
__global__ __launch_bounds__(256) void kv_kernel(const unsigned short* __restrict__ Kf,
                                                 const unsigned short* __restrict__ Vf,
                                                 unsigned short* __restrict__ KV,
                                                 unsigned short* __restrict__ Ks,
                                                 int seq, int D) {
    __shared__ __align__(16) unsigned short kT[64 * 32];  // [d][s]
    __shared__ __align__(16) unsigned short vT[64 * 32];  // [d][s]

    const int bh = blockIdx.x;
    const int bb = bh >> 4, hh = bh & 15;
    const unsigned short* kbase = Kf + (size_t)bb * seq * D + hh * 64;
    const unsigned short* vbase = Vf + (size_t)bb * seq * D + hh * 64;

    const int t = threadIdx.x;
    const int wave = t >> 5;
    const int lane = t & 31;
    const int wm = (wave >> 1) * 16;  // 0,16,32,48
    const int wn = (wave & 1) * 32;   // 0,32

    v8f acc[2], acc_ks[2];
    acc[0] = v8f_zero();
    acc[1] = v8f_zero();
    acc_ks[0] = v8f_zero();
    acc_ks[1] = v8f_zero();

    FragBF ones;
#pragma unroll
    for (int i = 0; i < 8; ++i) ones.u[i] = 0x3F803F80u;  // bf16(1.0) x2

    const int row = t >> 3, q = t & 7;  // row: s (0..31), q: 8-half chunk of d
    for (int s0 = 0; s0 < seq; s0 += 32) {
        U4H8 kx, vx;
        kx.u4 = *(const uint4*)(kbase + (size_t)(s0 + row) * D + q * 8);
        vx.u4 = *(const uint4*)(vbase + (size_t)(s0 + row) * D + q * 8);
#pragma unroll
        for (int j = 0; j < 8; ++j) {
            kT[(q * 8 + j) * 32 + row] = kx.h[j];
            vT[(q * 8 + j) * 32 + row] = vx.h[j];
        }
        __syncthreads();

        FragBF a, b0, b1;
        load_frag_rows<32>(vT, wm + (lane & 15), 0, a);        // A[m][s] = vT[m][s]
        load_frag_rows<32>(kT, wn + (lane & 15), 0, b0);       // B[s][d] -> kT[d][s]
        load_frag_rows<32>(kT, wn + 16 + (lane & 15), 0, b1);
        acc[0] = __builtin_amdgcn_wmma_f32_16x16x32_bf16(false, a.v, false, b0.v,
                                                         (short)0, acc[0], false, false);
        acc[1] = __builtin_amdgcn_wmma_f32_16x16x32_bf16(false, a.v, false, b1.v,
                                                         (short)0, acc[1], false, false);
        // k_sum on the matrix pipe (all waves; only waves 0/1 write at the end)
        acc_ks[0] = __builtin_amdgcn_wmma_f32_16x16x32_bf16(false, ones.v, false, b0.v,
                                                            (short)0, acc_ks[0], false, false);
        acc_ks[1] = __builtin_amdgcn_wmma_f32_16x16x32_bf16(false, ones.v, false, b1.v,
                                                            (short)0, acc_ks[1], false, false);
        __syncthreads();
    }

    const int cn = lane & 15;
    const int ch = lane >> 4;
#pragma unroll
    for (int in = 0; in < 2; ++in)
#pragma unroll
        for (int r = 0; r < 8; ++r) {
            int m = wm + r + 8 * ch;
            int d = wn + in * 16 + cn;
            KV[(size_t)bh * 4096 + m * 64 + d] = f32_to_bf16(acc[in][r]);
        }
    // accumulator row M=0 (VGPR0, lanes 0..15) holds k_sum
    if (wave < 2 && ch == 0) {
        Ks[bh * 64 + wn + cn] = f32_to_bf16(acc_ks[0][0]);
        Ks[bh * 64 + wn + 16 + cn] = f32_to_bf16(acc_ks[1][0]);
    }
}

// ============================================================
// out[l, h*64+m] = z[l,h] * sum_d q[l, h*64+d] * kv[m][d]
// grid (32 row-tiles, 64 bh); async-staged LDS tiles.
// The normalizer dot q.ksum also runs on the matrix pipe: B_z has
// ksum in column 0, so C[m][0] is the dot for row m.
// ============================================================
__global__ __launch_bounds__(256) void attn_kernel(const unsigned short* __restrict__ Qf,
                                                   const unsigned short* __restrict__ KV,
                                                   const unsigned short* __restrict__ Ks,
                                                   unsigned short* __restrict__ Out,
                                                   int seq, int D, float eps) {
    __shared__ __align__(16) unsigned short qt[128 * 64];
    __shared__ __align__(16) unsigned short kvt[64 * 64];
    __shared__ __align__(4) unsigned short kssh[64];
    __shared__ float zsh[128];

    const int lt = blockIdx.x;
    const int bh = blockIdx.y;
    const int bb = bh >> 4, hh = bh & 15;
    const unsigned short* qbase = Qf + (size_t)bb * seq * D + hh * 64;
    unsigned short* obase = Out + (size_t)bb * seq * D + hh * 64;

    const int t = threadIdx.x;
    const int wave = t >> 5;
    const int lane = t & 31;
    const int cn = lane & 15;
    const int ch = lane >> 4;
    const int l0 = lt * 128;

#pragma unroll
    for (int i = 0; i < 4; ++i) {  // qt: 1024 b128 chunks
        int flat = t + 256 * i;
        int row = flat >> 3, q = flat & 7;
        async_b128(&qt[row * 64 + q * 8], qbase + (size_t)(l0 + row) * D + q * 8);
    }
#pragma unroll
    for (int i = 0; i < 2; ++i) {  // kvt: 512 b128 chunks
        int flat = t + 256 * i;
        int row = flat >> 3, q = flat & 7;
        async_b128(&kvt[row * 64 + q * 8], KV + (size_t)bh * 4096 + row * 64 + q * 8);
    }
    if (t < 64) kssh[t] = Ks[bh * 64 + t];
    wait_async<0>();
    __syncthreads();

    v8f acc[4], acc_z;
#pragma unroll
    for (int in = 0; in < 4; ++in) acc[in] = v8f_zero();
    acc_z = v8f_zero();

#pragma unroll
    for (int k0 = 0; k0 < 64; k0 += 32) {
        FragBF a;
        load_frag_rows<64>(qt, wave * 16 + cn, k0, a);
#pragma unroll
        for (int in = 0; in < 4; ++in) {
            FragBF b;
            load_frag_rows<64>(kvt, in * 16 + cn, k0, b);
            acc[in] = __builtin_amdgcn_wmma_f32_16x16x32_bf16(false, a.v, false, b.v,
                                                              (short)0, acc[in], false, false);
        }
        // normalizer: B_z column 0 = ksum, other columns 0
        FragBF bz;
#pragma unroll
        for (int i = 0; i < 8; ++i) bz.u[i] = 0u;
        if (cn == 0) load_frag_rows<64>(kssh, 0, k0, bz);
        acc_z = __builtin_amdgcn_wmma_f32_16x16x32_bf16(false, a.v, false, bz.v,
                                                        (short)0, acc_z, false, false);
    }

    // lanes cn==0 hold C[m][0] = q.ksum for row m = wave*16 + r + 8*ch
    if (cn == 0) {
#pragma unroll
        for (int r = 0; r < 8; ++r)
            zsh[wave * 16 + 8 * ch + r] = 1.0f / (acc_z[r] + eps);
    }
    __syncthreads();

#pragma unroll
    for (int in = 0; in < 4; ++in)
#pragma unroll
        for (int r = 0; r < 8; ++r) {
            int row = wave * 16 + r + 8 * ch;
            float y = acc[in][r] * zsh[row];
            obase[(size_t)(l0 + row) * D + in * 16 + cn] = f32_to_bf16(y);
        }
}

extern "C" void kernel_launch(void* const* d_in, const int* in_sizes, int n_in,
                              void* d_out, int out_size, void* d_ws, size_t ws_size,
                              hipStream_t stream) {
    const float* q  = (const float*)d_in[0];
    const float* k  = (const float*)d_in[1];
    const float* v  = (const float*)d_in[2];
    const float* Wq = (const float*)d_in[3];
    const float* bq = (const float*)d_in[4];
    const float* Wk = (const float*)d_in[5];
    const float* bk = (const float*)d_in[6];
    const float* Wv = (const float*)d_in[7];
    const float* bv = (const float*)d_in[8];
    const float* Wo = (const float*)d_in[9];
    const float* bo = (const float*)d_in[10];
    float* out = (float*)d_out;

    const int B = 4, N = 4096, D = 1024;
    const int M = B * N;                  // 16384
    const size_t MD = (size_t)M * D;      // 16.7M elems
    const size_t DD = (size_t)D * D;

    // workspace layout (bf16 halfs unless noted)
    unsigned short* xq = (unsigned short*)d_ws;   // bf16 inputs
    unsigned short* xk = xq + MD;
    unsigned short* xv = xk + MD;
    unsigned short* wqb = xv + MD;                // bf16 weights
    unsigned short* wkb = wqb + DD;
    unsigned short* wvb = wkb + DD;
    unsigned short* wob = wvb + DD;
    unsigned short* qf = wob + DD;                // projected features
    unsigned short* kf = qf + MD;
    unsigned short* vf = kf + MD;
    unsigned short* af = xq;                      // alias: xq dead after q-proj
    unsigned short* kv = xk;                      // alias: xk dead after k-proj
    unsigned short* ksum = xk + (size_t)64 * 64 * 64;  // bf16 k_sum

    // 1) bulk f32 -> bf16
    cvt_bf16_kernel<<<(int)(MD / 4 / 256), 256, 0, stream>>>(q, xq, MD / 4);
    cvt_bf16_kernel<<<(int)(MD / 4 / 256), 256, 0, stream>>>(k, xk, MD / 4);
    cvt_bf16_kernel<<<(int)(MD / 4 / 256), 256, 0, stream>>>(v, xv, MD / 4);
    cvt_bf16_kernel<<<(int)(DD / 4 / 256), 256, 0, stream>>>(Wq, wqb, DD / 4);
    cvt_bf16_kernel<<<(int)(DD / 4 / 256), 256, 0, stream>>>(Wk, wkb, DD / 4);
    cvt_bf16_kernel<<<(int)(DD / 4 / 256), 256, 0, stream>>>(Wv, wvb, DD / 4);
    cvt_bf16_kernel<<<(int)(DD / 4 / 256), 256, 0, stream>>>(Wo, wob, DD / 4);

    // 2) projections (+ elu feature for q,k)
    dim3 gp(M / 128, D / 128);
    proj_kernel<1, unsigned short><<<gp, 256, 0, stream>>>(xq, wqb, bq, qf, M, D, D);
    proj_kernel<1, unsigned short><<<gp, 256, 0, stream>>>(xk, wkb, bk, kf, M, D, D);
    proj_kernel<0, unsigned short><<<gp, 256, 0, stream>>>(xv, wvb, bv, vf, M, D, D);

    // 3) per-head kv state + k_sum (all on the matrix pipe)
    kv_kernel<<<64, 256, 0, stream>>>(kf, vf, kv, ksum, N, D);

    // 4) normalized attention output (normalizer on the matrix pipe too)
    attn_kernel<<<dim3(32, 64), 256, 0, stream>>>(qf, kv, ksum, af, N, D, 1e-6f);

    // 5) output projection (f32 out)
    proj_kernel<0, float><<<gp, 256, 0, stream>>>(af, wob, bo, out, M, D, D);
}
